// TextOverRegionAttention_61048665145682
// MI455X (gfx1250) — matrix operand order repaired and checked by reading, not compiled, not run on hardware
//
#include <hip/hip_runtime.h>
#include <cstdint>
#include <cstddef>

// ---------------- problem sizes (fixed by reference) ----------------
static constexpr int Bz = 4, Lq = 4096, Nr = 1024, Dm = 768, Hn = 12, HDm = 64, Rr = 8;
static constexpr float LORA_SCALE = 2.0f;     // 16/8
static constexpr float EPS_LN     = 1e-5f;
static constexpr float ATT_SCALE  = 0.125f;   // 1/sqrt(64)
static constexpr float NEG_BIG    = -3.0e38f;

typedef __attribute__((ext_vector_type(16))) __bf16     v16bf;
typedef __attribute__((ext_vector_type(8)))  float      v8f;
typedef __attribute__((ext_vector_type(8)))  uint32_t   v8u;

#define DEV __device__ __forceinline__

DEV uint16_t f2bf(float x) {
  uint32_t u = __builtin_bit_cast(uint32_t, x);
  uint32_t r = (u + 0x7FFFu + ((u >> 16) & 1u)) >> 16;   // round-to-nearest-even
  return (uint16_t)r;
}
DEV float bf2f(uint16_t h) {
  uint32_t u = ((uint32_t)h) << 16;
  return __builtin_bit_cast(float, u);
}

// ---------------- elementwise f32 -> bf16 ----------------
__global__ void cvt_f32_bf16_kernel(const float* __restrict__ in,
                                    uint16_t* __restrict__ out, int n) {
  int i = blockIdx.x * 256 + threadIdx.x;
  if (i < n) out[i] = f2bf(in[i]);
}

// ---------------- fused LayerNorm (row = 768) -> bf16 ----------------
__global__ void ln_bf16_kernel(const float* __restrict__ x,
                               const float* __restrict__ gamma,
                               const float* __restrict__ beta,
                               uint16_t* __restrict__ out) {
  int row = blockIdx.x;
  const float* xr = x + (size_t)row * Dm;
  int t = threadIdx.x;                       // 256 threads, 3 elems each
  float v0 = xr[t], v1 = xr[t + 256], v2 = xr[t + 512];
  float s  = v0 + v1 + v2;
  float sq = v0 * v0 + v1 * v1 + v2 * v2;
  #pragma unroll
  for (int m = 1; m < 32; m <<= 1) {
    s  += __shfl_xor(s,  m, 32);
    sq += __shfl_xor(sq, m, 32);
  }
  __shared__ float ss[8], sqq[8];
  int wid = t >> 5, lid = t & 31;
  if (lid == 0) { ss[wid] = s; sqq[wid] = sq; }
  __syncthreads();
  s = 0.f; sq = 0.f;
  #pragma unroll
  for (int w = 0; w < 8; w++) { s += ss[w]; sq += sqq[w]; }
  float mu   = s * (1.0f / Dm);
  float var  = sq * (1.0f / Dm) - mu * mu;
  float rstd = rsqrtf(var + EPS_LN);
  uint16_t* orow = out + (size_t)row * Dm;
  orow[t]       = f2bf((v0 - mu) * rstd * gamma[t]       + beta[t]);
  orow[t + 256] = f2bf((v1 - mu) * rstd * gamma[t + 256] + beta[t + 256]);
  orow[t + 512] = f2bf((v2 - mu) * rstd * gamma[t + 512] + beta[t + 512]);
}

// ---------------- LoRA down-proj: U[row, r] = sum_k X[row,k] * A[r,k] ----------------
__global__ void lora_down_kernel(const uint16_t* __restrict__ xbf,
                                 const float* __restrict__ A,
                                 float* __restrict__ U, int rows) {
  int gw   = (blockIdx.x * blockDim.x + threadIdx.x) >> 5;
  int lane = threadIdx.x & 31;
  if (gw >= rows) return;
  const uint16_t* xr = xbf + (size_t)gw * Dm;
  float acc[Rr];
  #pragma unroll
  for (int r = 0; r < Rr; r++) acc[r] = 0.f;
  for (int k = lane; k < Dm; k += 32) {
    float xv = bf2f(xr[k]);
    #pragma unroll
    for (int r = 0; r < Rr; r++) acc[r] += xv * A[r * Dm + k];
  }
  #pragma unroll
  for (int r = 0; r < Rr; r++) {
    #pragma unroll
    for (int m = 1; m < 32; m <<= 1) acc[r] += __shfl_xor(acc[r], m, 32);
  }
  if (lane == 0) {
    #pragma unroll
    for (int r = 0; r < Rr; r++) U[(size_t)gw * Rr + r] = acc[r];
  }
}

// ---------------- WMMA bf16 GEMM: D = alpha*(A*W^T + 2*U*Bl^T) (+bias)(+resid) ----------
// A: (M,K) bf16 row-major; W: (Nout,K) bf16 row-major ("torch" (out,in) layout).
// Wave tile: 32(M) x 64(N); block = 8 waves = 64(M) x 256(N).
template <bool OUT_BF16, bool LORA, bool RESID>
__global__ __launch_bounds__(256, 1)
void gemm_wmma_kernel(const uint16_t* __restrict__ Abf,
                      const uint16_t* __restrict__ Wbf,
                      const float* __restrict__ U,       // (M,R)
                      const float* __restrict__ Blora,   // (Nout,R)
                      const float* __restrict__ bias,    // (Nout) or null
                      const float* __restrict__ resid,   // (M,Nout) or null
                      void* __restrict__ out,
                      int M, int Nout, int K, float alpha) {
  int wv    = threadIdx.x >> 5;
  int lane  = threadIdx.x & 31;
  int lhalf = lane & 15, g = lane >> 4;
  int m0 = blockIdx.y * 64 + (wv >> 2) * 32;
  int n0 = blockIdx.x * 256 + (wv & 3) * 64;

  v8f acc[2][4] = {};
  const uint16_t* Arow0 = Abf + (size_t)(m0 + lhalf) * K;
  const uint16_t* Arow1 = Arow0 + (size_t)16 * K;

  for (int k0 = 0; k0 < K; k0 += 32) {
    v8u au0, au1;
    #pragma unroll
    for (int v = 0; v < 8; v++) {   // A layout: lanes 0-15 K=0..15, 16-31 K=16..31
      int kk = k0 + 16 * g + 2 * v;
      au0[v] = *(const uint32_t*)(Arow0 + kk);
      au1[v] = *(const uint32_t*)(Arow1 + kk);
    }
    v16bf af0 = __builtin_bit_cast(v16bf, au0);
    v16bf af1 = __builtin_bit_cast(v16bf, au1);
    #pragma unroll
    for (int t = 0; t < 4; t++) {
      const uint16_t* Wrow = Wbf + (size_t)(n0 + t * 16 + lhalf) * K;
      v8u bu;
      #pragma unroll
      for (int v = 0; v < 8; v++) { // B layout, same k order as A: lanes 0-15 K=0..15, 16-31 K=16..31
        int kk = k0 + 16 * g + 2 * v;
        bu[v] = *(const uint32_t*)(Wrow + kk);
      }
      v16bf bfg = __builtin_bit_cast(v16bf, bu);
      acc[0][t] = __builtin_amdgcn_wmma_f32_16x16x32_bf16(
                      false, af0, false, bfg, (short)0, acc[0][t], false, false);
      acc[1][t] = __builtin_amdgcn_wmma_f32_16x16x32_bf16(
                      false, af1, false, bfg, (short)0, acc[1][t], false, false);
    }
  }

  #pragma unroll
  for (int t = 0; t < 4; t++) {
    int ncol = n0 + t * 16 + lhalf;
    float bl[Rr];
    if (LORA) {
      #pragma unroll
      for (int r = 0; r < Rr; r++) bl[r] = Blora[(size_t)ncol * Rr + r];
    }
    float bb = bias ? bias[ncol] : 0.0f;
    #pragma unroll
    for (int mt = 0; mt < 2; mt++) {
      #pragma unroll
      for (int v = 0; v < 8; v++) {
        int mrow  = m0 + mt * 16 + v + 8 * g;
        float val = acc[mt][t][v];
        if (LORA) {
          const float* ur = U + (size_t)mrow * Rr;
          float lsum = 0.f;
          #pragma unroll
          for (int r = 0; r < Rr; r++) lsum += ur[r] * bl[r];
          val += LORA_SCALE * lsum;
        }
        val = val * alpha + bb;
        size_t oidx = (size_t)mrow * Nout + ncol;
        if (RESID) val += resid[oidx];
        if (OUT_BF16) ((uint16_t*)out)[oidx] = f2bf(val);
        else          ((float*)out)[oidx]    = val;
      }
    }
  }
}

// ---------------- V transpose: (B,N,H,HD) -> (B,H,HD,N) bf16 ----------------
__global__ void transpose_v_kernel(const uint16_t* __restrict__ V,
                                   uint16_t* __restrict__ Vt) {
  int i = blockIdx.x * 256 + threadIdx.x;
  int total = Bz * Nr * Dm;
  if (i >= total) return;
  int d = i % Dm;
  int n = (i / Dm) % Nr;
  int b = i / (Dm * Nr);
  int h = d / HDm, dd = d % HDm;
  Vt[((size_t)(b * Hn + h) * HDm + dd) * Nr + n] = V[i];
}

// ---------------- flash attention: one wave per (b,h,16-row L tile) ----------------
// Q already scaled by 1/sqrt(HD). Q,K bf16 (row-major, D stride); Vt bf16 (B,H,HD,N).
__global__ __launch_bounds__(256, 1)
void attn_wmma_kernel(const uint16_t* __restrict__ Qb,
                      const uint16_t* __restrict__ Kb,
                      const uint16_t* __restrict__ Vt,
                      const int* __restrict__ mask,
                      uint16_t* __restrict__ Ob) {
  __shared__ uint16_t lds_p[8][16 * 64];        // per-wave 2KB P-transpose patch
  int wv    = threadIdx.x >> 5;
  int lane  = threadIdx.x & 31;
  int lhalf = lane & 15, g = lane >> 4;
  int gw    = blockIdx.x * 8 + wv;
  int ltile = gw & 255;                          // Lq/16 == 256 tiles per (b,h)
  int bh    = gw >> 8;
  int h     = bh % Hn;
  int b     = bh / Hn;
  int l0    = ltile * 16;

  // loop-invariant bases
  const int*      maskb = mask + b * Nr;
  const uint16_t* Kbase = Kb + (size_t)(b * Nr) * Dm + h * HDm;
  const uint16_t* Vbase = Vt + (size_t)((b * Hn + h) * HDm + lhalf) * Nr;

  // resident Q A-fragments (16 x 64 over d, as 2 x K=32 frags)
  const uint16_t* Qrow = Qb + (size_t)(b * Lq + l0 + lhalf) * Dm + h * HDm;
  v16bf qa[2];
  #pragma unroll
  for (int kt = 0; kt < 2; kt++) {
    v8u qu;
    #pragma unroll
    for (int v = 0; v < 8; v++) {
      int d = 32 * kt + 16 * g + 2 * v;
      qu[v] = *(const uint32_t*)(Qrow + d);
    }
    qa[kt] = __builtin_bit_cast(v16bf, qu);
  }

  float mi[8], li[8];
  v8f Oacc[4] = {};
  #pragma unroll
  for (int v = 0; v < 8; v++) { mi[v] = NEG_BIG; li[v] = 0.f; }
  uint16_t* Pl = lds_p[wv];

  for (int nt = 0; nt < Nr; nt += 64) {
    // ---- S = (Q * 1/8) @ K^T over a 16 x 64 tile ----
    v8f  S[4];
    bool mk[4];
    #pragma unroll
    for (int t = 0; t < 4; t++) {
      int nn = nt + t * 16 + lhalf;
      mk[t]  = maskb[nn] != 0;
      const uint16_t* Krow = Kbase + (size_t)nn * Dm;
      v8f s = {};
      #pragma unroll
      for (int kt = 0; kt < 2; kt++) {
        v8u ku;
        #pragma unroll
        for (int v = 0; v < 8; v++) {
          int d = 32 * kt + 16 * g + 2 * v;     // B-matrix K layout
          ku[v] = *(const uint32_t*)(Krow + d);
        }
        s = __builtin_amdgcn_wmma_f32_16x16x32_bf16(
                false, qa[kt], false, __builtin_bit_cast(v16bf, ku),
                (short)0, s, false, false);
      }
      S[t] = s;
    }
    // ---- online softmax row stats (rows live per-VGPR, cols per-lane half) ----
    float fac[8];
    #pragma unroll
    for (int v = 0; v < 8; v++) {
      float mx = NEG_BIG;
      #pragma unroll
      for (int t = 0; t < 4; t++) if (mk[t]) mx = fmaxf(mx, S[t][v]);
      #pragma unroll
      for (int m = 1; m < 16; m <<= 1) mx = fmaxf(mx, __shfl_xor(mx, m, 32));
      float mnew = fmaxf(mi[v], mx);
      fac[v] = __expf(mi[v] - mnew);            // exp(-huge)->0, exp(0)->1: NaN-free
      float rs = 0.f;
      #pragma unroll
      for (int t = 0; t < 4; t++) {
        float pv = mk[t] ? __expf(S[t][v] - mnew) : 0.0f;
        S[t][v] = pv;
        rs += pv;
      }
      #pragma unroll
      for (int m = 1; m < 16; m <<= 1) rs += __shfl_xor(rs, m, 32);
      li[v] = li[v] * fac[v] + rs;
      mi[v] = mnew;
    }
    #pragma unroll
    for (int t2 = 0; t2 < 4; t2++)
      #pragma unroll
      for (int v = 0; v < 8; v++) Oacc[t2][v] *= fac[v];

    // ---- P: C-layout -> A-layout via same-wave LDS round trip ----
    #pragma unroll
    for (int t = 0; t < 4; t++)
      #pragma unroll
      for (int v = 0; v < 8; v++)
        Pl[(v + 8 * g) * 64 + t * 16 + lhalf] = f2bf(S[t][v]);

    v16bf pa[2];
    #pragma unroll
    for (int kt = 0; kt < 2; kt++) {
      v8u pu;
      #pragma unroll
      for (int v = 0; v < 8; v++) {
        int kk = 32 * kt + 16 * g + 2 * v;
        pu[v] = *(const uint32_t*)(Pl + lhalf * 64 + kk);
      }
      pa[kt] = __builtin_bit_cast(v16bf, pu);
    }
    // ---- O += P @ V  (B-frags contiguous thanks to Vt layout) ----
    #pragma unroll
    for (int kt = 0; kt < 2; kt++) {
      #pragma unroll
      for (int t2 = 0; t2 < 4; t2++) {
        const uint16_t* Vrow = Vbase + (size_t)(t2 * 16) * Nr;
        v8u vu;
        #pragma unroll
        for (int v = 0; v < 8; v++) {
          int kk = nt + 32 * kt + 16 * g + 2 * v;
          vu[v] = *(const uint32_t*)(Vrow + kk);
        }
        Oacc[t2] = __builtin_amdgcn_wmma_f32_16x16x32_bf16(
                       false, pa[kt], false, __builtin_bit_cast(v16bf, vu),
                       (short)0, Oacc[t2], false, false);
      }
    }
  }

  // ---- normalize (fully-masked rows -> 0, mirrors nan_to_num) and store bf16 ----
  float inv[8];
  #pragma unroll
  for (int v = 0; v < 8; v++) inv[v] = (li[v] > 0.f) ? (1.0f / li[v]) : 0.0f;
  #pragma unroll
  for (int t2 = 0; t2 < 4; t2++)
    #pragma unroll
    for (int v = 0; v < 8; v++) {
      int mrow = l0 + v + 8 * g;
      Ob[(size_t)(b * Lq + mrow) * Dm + h * HDm + t2 * 16 + lhalf] =
          f2bf(Oacc[t2][v] * inv[v]);
    }
}

// ---------------- host orchestration ----------------
extern "C" void kernel_launch(void* const* d_in, const int* in_sizes, int n_in,
                              void* d_out, int out_size, void* d_ws, size_t ws_size,
                              hipStream_t stream) {
  (void)in_sizes; (void)n_in; (void)out_size; (void)ws_size;
  const float* text    = (const float*)d_in[0];
  const float* regions = (const float*)d_in[1];
  const int*   mask    = (const int*)  d_in[2];
  const float* Wq = (const float*)d_in[3];
  const float* Aq = (const float*)d_in[4];
  const float* Bq = (const float*)d_in[5];
  const float* Wk = (const float*)d_in[6];
  const float* Wv = (const float*)d_in[7];
  const float* Av = (const float*)d_in[8];
  const float* Bv = (const float*)d_in[9];
  const float* Wo = (const float*)d_in[10];
  const float* bo = (const float*)d_in[11];
  const float* gq  = (const float*)d_in[12];
  const float* bqn = (const float*)d_in[13];
  const float* gkv = (const float*)d_in[14];
  const float* bkv = (const float*)d_in[15];

  char* p = (char*)d_ws;
  auto alloc = [&](size_t bytes) -> char* {
    char* r = p;
    p += (bytes + 255) & ~(size_t)255;
    return r;
  };
  const size_t ML = (size_t)Bz * Lq;   // 16384 text rows
  const size_t MN = (size_t)Bz * Nr;   // 4096 region rows
  uint16_t* tq    = (uint16_t*)alloc(ML * Dm * 2);
  uint16_t* rk    = (uint16_t*)alloc(MN * Dm * 2);
  uint16_t* regbf = (uint16_t*)alloc(MN * Dm * 2);
  uint16_t* wqb   = (uint16_t*)alloc((size_t)Dm * Dm * 2);
  uint16_t* wkb   = (uint16_t*)alloc((size_t)Dm * Dm * 2);
  uint16_t* wvb   = (uint16_t*)alloc((size_t)Dm * Dm * 2);
  uint16_t* wob   = (uint16_t*)alloc((size_t)Dm * Dm * 2);
  float*    Uq    = (float*)   alloc(ML * Rr * 4);
  float*    Uv    = (float*)   alloc(MN * Rr * 4);
  uint16_t* Qb    = (uint16_t*)alloc(ML * Dm * 2);
  uint16_t* Kbuf  = (uint16_t*)alloc(MN * Dm * 2);
  uint16_t* Vbuf  = (uint16_t*)alloc(MN * Dm * 2);
  uint16_t* Vt    = (uint16_t*)alloc(MN * Dm * 2);
  uint16_t* AO    = (uint16_t*)alloc(ML * Dm * 2);

  const int nW = Dm * Dm;
  cvt_f32_bf16_kernel<<<(nW + 255) / 256, 256, 0, stream>>>(Wq, wqb, nW);
  cvt_f32_bf16_kernel<<<(nW + 255) / 256, 256, 0, stream>>>(Wk, wkb, nW);
  cvt_f32_bf16_kernel<<<(nW + 255) / 256, 256, 0, stream>>>(Wv, wvb, nW);
  cvt_f32_bf16_kernel<<<(nW + 255) / 256, 256, 0, stream>>>(Wo, wob, nW);

  ln_bf16_kernel<<<(int)ML, 256, 0, stream>>>(text,    gq,  bqn, tq);
  ln_bf16_kernel<<<(int)MN, 256, 0, stream>>>(regions, gkv, bkv, rk);
  {
    int nR = (int)(MN * Dm);
    cvt_f32_bf16_kernel<<<(nR + 255) / 256, 256, 0, stream>>>(regions, regbf, nR);
  }

  lora_down_kernel<<<(int)(ML / 8), 256, 0, stream>>>(tq,    Aq, Uq, (int)ML);
  lora_down_kernel<<<(int)(MN / 8), 256, 0, stream>>>(regbf, Av, Uv, (int)MN);

  // Q = 1/8 * (tq@Wq^T + 2*Uq@Bq^T)   (scale folded in; feeds attention directly)
  gemm_wmma_kernel<true, true, false>
      <<<dim3(Dm / 256, (int)(ML / 64)), 256, 0, stream>>>(
          tq, wqb, Uq, Bq, nullptr, nullptr, Qb, (int)ML, Dm, Dm, ATT_SCALE);
  // K = rk@Wk^T
  gemm_wmma_kernel<true, false, false>
      <<<dim3(Dm / 256, (int)(MN / 64)), 256, 0, stream>>>(
          rk, wkb, nullptr, nullptr, nullptr, nullptr, Kbuf, (int)MN, Dm, Dm, 1.0f);
  // V = regions@Wv^T + 2*Uv@Bv^T
  gemm_wmma_kernel<true, true, false>
      <<<dim3(Dm / 256, (int)(MN / 64)), 256, 0, stream>>>(
          regbf, wvb, Uv, Bv, nullptr, nullptr, Vbuf, (int)MN, Dm, Dm, 1.0f);

  {
    int nV = (int)(MN * Dm);
    transpose_v_kernel<<<(nV + 255) / 256, 256, 0, stream>>>(Vbuf, Vt);
  }

  // flash attention: B*H*(L/16) waves / 8 waves per block
  attn_wmma_kernel<<<(Bz * Hn * (Lq / 16)) / 8, 256, 0, stream>>>(
      Qb, Kbuf, Vt, mask, AO);

  // out = text + AO@Wo^T + bo  (f32 output)
  gemm_wmma_kernel<false, false, true>
      <<<dim3(Dm / 256, (int)(ML / 64)), 256, 0, stream>>>(
          AO, wob, nullptr, nullptr, bo, text, d_out, (int)ML, Dm, Dm, 1.0f);
}
